// MambaBlock_40037685133975
// MI455X (gfx1250) — matrix operand order repaired and checked by reading, not compiled, not run on hardware
//
#include <hip/hip_runtime.h>
#include <math.h>

// ---------------- problem constants ----------------
#define BB 2
#define TT 6
#define PP 20480
#define DIMC 96
#define DIMS 8
#define INHRES 28
#define EMBH 32
#define DINNER 128
#define NHEADS 8
#define CONVDIM 130
#define DINPROJ 266
#define GG0 3
#define GG1 10
#define LSEQ 30
#define SEQ_PER_WG 8
#define TOK_PER_WG 240          // 15 WMMA M-tiles of 16
#define NSEQ (BB*(TT/GG0)*(PP/GG1))   // 8192
#define NWG (NSEQ/SEQ_PER_WG)         // 1024
#define EPSV 1e-5f

// ---------------- WMMA types ----------------
typedef __bf16 v16bf __attribute__((ext_vector_type(16)));
typedef float  v8f   __attribute__((ext_vector_type(8)));

union FragU { v16bf v; unsigned int u[8]; };

__device__ __forceinline__ unsigned short f2bf(float f) {
    union { float f; unsigned int u; } c; c.f = f;
    unsigned int u = c.u;
    unsigned int r = (u + 0x7fffu + ((u >> 16) & 1u)) >> 16;
    return (unsigned short)r;
}
__device__ __forceinline__ float bf2f(unsigned short h) {
    union { unsigned int u; float f; } c; c.u = ((unsigned int)h) << 16;
    return c.f;
}
__device__ __forceinline__ float sigm_(float x) { return 1.f / (1.f + __expf(-x)); }
__device__ __forceinline__ float silu_(float x) { return x * sigm_(x); }
__device__ __forceinline__ float softplus_(float x) { return (x > 20.f) ? x : log1pf(__expf(x)); }

// A fragment (16x32 bf16) from LDS, row-major with given stride (in halves).
// 16-bit A layout (ISA 7.12.2): lane m=lane&15, hi=lane>>4;
// dword v holds K pair at kb = (v<4?0:16) + hi*8 + (v&3)*2.
// (bytes: two contiguous 16B chunks -> compiler emits 2x ds_load_b128)
__device__ __forceinline__ v16bf load_afrag(const unsigned short* sbase,
                                            int m0, int k0, int stride, int lane) {
    FragU f;
    int m = m0 + (lane & 15);
    int hi = lane >> 4;
    const unsigned short* row = sbase + m * stride + k0;
#pragma unroll
    for (int v = 0; v < 8; ++v) {
        int kb = ((v < 4) ? 0 : 16) + hi * 8 + (v & 3) * 2;
        f.u[v] = *(const unsigned int*)(row + kb);
    }
    return f.v;
}

// B fragment prepacked in global memory: [frag][lane][16 halves] contiguous.
__device__ __forceinline__ v16bf load_bfrag(const unsigned short* g, int frag, int lane) {
    FragU f;
    const unsigned int* p = (const unsigned int*)(g + ((size_t)frag * 32 + lane) * 16);
#pragma unroll
    for (int i = 0; i < 8; ++i) f.u[i] = p[i];
    return f.v;
}

__device__ __forceinline__ v8f wmma_bf16(v16bf a, v16bf b, v8f c) {
    return __builtin_amdgcn_wmma_f32_16x16x32_bf16(false, a, false, b, (short)0, c, false, false);
}

__device__ __forceinline__ void tok2btp(int blk, int tok, int& b, int& t, int& p) {
    int s = tok / LSEQ, l = tok % LSEQ;
    int g0 = l / GG1, g1 = l % GG1;
    int seq = blk * SEQ_PER_WG + s;
    b = seq >> 12;
    int tg = (seq >> 11) & 1;
    int pg = seq & 2047;
    t = tg * GG0 + g0;
    p = pg * GG1 + g1;
}
__device__ __forceinline__ int tok_ord(int s, int l, int dir) {
    int g0 = l / GG1, q = l % GG1;
    int g1 = dir ? (9 - q) : q;
    return s * LSEQ + g0 * GG1 + g1;
}

// -------- weight pre-pack: f32 [K][N] -> bf16 B fragments --------
// B layout (ISA 7.12.2, dense 32x16 16-bit): lane n=lane&15, hi=lane>>4,
// element e -> K = kt*32 + (e<8 ? hi*8 + e : 16 + hi*8 + (e-8)).
__global__ void pack_wmma_b(const float* __restrict__ w, unsigned short* __restrict__ dst,
                            int K, int N, int KT) {
    int frag = blockIdx.x;          // frag = nt*KT + kt
    int nt = frag / KT, kt = frag % KT;
    int lane = threadIdx.x;
    int hi = lane >> 4;
    int n = nt * 16 + (lane & 15);
    size_t base = ((size_t)frag * 32 + lane) * 16;
#pragma unroll
    for (int e = 0; e < 16; ++e) {
        int k = kt * 32 + ((e < 8) ? (hi * 8 + e) : (16 + hi * 8 + (e - 8)));
        unsigned short v = 0;
        if (k < K && n < N) v = f2bf(w[k * N + n]);
        dst[base + e] = v;
    }
}

// ---------------- fused forward ----------------
// LDS plan (bytes):
//  sA   @ 0      : 240*128 bf16 = 61440   (xn -> y_avg -> z2)
//  sXS  @ 61440  : 240*8  f32  = 7680     (gathered x_static)
//  pool @ 69120  : 211200
//    phase A  : sXC f32 240*128 = 122880
//    scan     : sZ bf16 61440 | sB bf16 240*132=63360 | sDT f32 7680 |
//               sdtA f32 15360 | sBC f32 1920 | sY bf16 61440
//    late     : sX1 f32 240*96=92160 | sH1 bf16 240*96=46080
#define SMEM_BYTES (61440 + 7680 + 211200)

__global__ __launch_bounds__(256, 2)
void fused_mamba_block(
    const float* __restrict__ x, const float* __restrict__ xs_g, const float* __restrict__ xh_g,
    const int* __restrict__ curves,
    const float* __restrict__ embed_w, const float* __restrict__ embed_b,
    const float* __restrict__ loan1_w, const float* __restrict__ loan1_b,
    const float* __restrict__ loan2_w, const float* __restrict__ loan2_b,
    const float* __restrict__ conv_w_f, const float* __restrict__ conv_b_f,
    const float* __restrict__ dt_bias_f, const float* __restrict__ A_log_f,
    const float* __restrict__ D_f, const float* __restrict__ norm_w_f,
    const float* __restrict__ conv_w_b, const float* __restrict__ conv_b_b,
    const float* __restrict__ dt_bias_b, const float* __restrict__ A_log_b,
    const float* __restrict__ D_b, const float* __restrict__ norm_w_b,
    const float* __restrict__ mlp_b1, const float* __restrict__ mlp_b2,
    const unsigned short* __restrict__ wIn, const unsigned short* __restrict__ wOut,
    const unsigned short* __restrict__ wM1, const unsigned short* __restrict__ wM2,
    float* __restrict__ out)
{
    __shared__ __align__(16) char SMEM[SMEM_BYTES];
    unsigned short* sA  = (unsigned short*)SMEM;
    float*          sXS = (float*)(SMEM + 61440);
    char*           pool = SMEM + 69120;

    float*          sXC = (float*)pool;                      // phase A
    unsigned short* sZ  = (unsigned short*)pool;             // scan
    unsigned short* sB  = (unsigned short*)(pool + 61440);
    float*          sDT = (float*)(pool + 124800);
    float*          sdtA = (float*)(pool + 132480);
    float*          sBC  = (float*)(pool + 147840);
    unsigned short* sY  = (unsigned short*)(pool + 149760);
    float*          sX1 = (float*)pool;                      // late
    unsigned short* sH1 = (unsigned short*)(pool + 92160);

    const int tid  = threadIdx.x;
    const int lane = tid & 31;
    const int wave = tid >> 5;
    const int blk  = blockIdx.x;

    // ---------- Phase A: gather + embed + LN(xc) + tanh(xs@loan1) -> sA (bf16) ----------
    if (tid < TOK_PER_WG) {
        int b, t, p; tok2btp(blk, tid, b, t, p);
        int psrc = curves[b * PP + p];
        long long base = ((long long)(b * TT + t) * PP + psrc);
        const float* xr  = x    + base * DIMC;
        const float* xsr = xs_g + base * DIMS;
        const float* xhr = xh_g + base * INHRES;

        float xs_l[8];
#pragma unroll
        for (int j = 0; j < 8; ++j) { xs_l[j] = xsr[j]; sXS[tid * 8 + j] = xs_l[j]; }
        float xh_l[28];
#pragma unroll
        for (int j = 0; j < 28; ++j) xh_l[j] = xhr[j];

        float sum = 0.f, sq = 0.f;
        for (int j = 0; j < DIMC; ++j) {
            float v = xr[j];
            sXC[tid * 128 + j] = v;
            sum += v; sq += v * v;
        }
        for (int j = 0; j < EMBH; ++j) {
            float a = embed_b[j];
            for (int k = 0; k < INHRES; ++k) a += xh_l[k] * embed_w[k * EMBH + j];
            a = tanhf(a);
            sXC[tid * 128 + DIMC + j] = a;
            sum += a; sq += a * a;
        }
        float mu = sum * (1.f / 128.f);
        float var = sq * (1.f / 128.f) - mu * mu;
        float rstd = rsqrtf(var + EPSV);
        for (int j = 0; j < DINNER; ++j) {
            float a = loan1_b[j];
            for (int k = 0; k < DIMS; ++k) a += xs_l[k] * loan1_w[k * DINNER + j];
            a = tanhf(a);
            float v = (sXC[tid * 128 + j] - mu) * rstd + a;
            sA[tid * 128 + j] = f2bf(v);
        }
    }
    __syncthreads();

    // ---------- in_proj GEMM: [240x128] @ [128x266] (pad N->272), 15x17 tiles ----------
    // nt loop split into uniform store regions: 0..7 -> z, 8..15 -> xBC[0..127],
    // 16 -> {xBC[128..129], dt, pad} (only divergent tile).
#pragma unroll 1
    for (int mt = wave; mt < 15; mt += 8) {
        v16bf a0 = load_afrag(sA, mt * 16,  0, 128, lane);
        v16bf a1 = load_afrag(sA, mt * 16, 32, 128, lane);
        v16bf a2 = load_afrag(sA, mt * 16, 64, 128, lane);
        v16bf a3 = load_afrag(sA, mt * 16, 96, 128, lane);
        const int nl    = lane & 15;
        const int mbase = mt * 16 + (lane >> 4) * 8;

#pragma unroll 1
        for (int nt = 0; nt < 8; ++nt) {           // z columns 0..127
            v8f acc = {};
            acc = wmma_bf16(a0, load_bfrag(wIn, nt * 4 + 0, lane), acc);
            acc = wmma_bf16(a1, load_bfrag(wIn, nt * 4 + 1, lane), acc);
            acc = wmma_bf16(a2, load_bfrag(wIn, nt * 4 + 2, lane), acc);
            acc = wmma_bf16(a3, load_bfrag(wIn, nt * 4 + 3, lane), acc);
            int n = nt * 16 + nl;
#pragma unroll
            for (int r = 0; r < 8; ++r) sZ[(mbase + r) * 128 + n] = f2bf(acc[r]);
        }
#pragma unroll 1
        for (int nt = 8; nt < 16; ++nt) {          // xBC columns 0..127
            v8f acc = {};
            acc = wmma_bf16(a0, load_bfrag(wIn, nt * 4 + 0, lane), acc);
            acc = wmma_bf16(a1, load_bfrag(wIn, nt * 4 + 1, lane), acc);
            acc = wmma_bf16(a2, load_bfrag(wIn, nt * 4 + 2, lane), acc);
            acc = wmma_bf16(a3, load_bfrag(wIn, nt * 4 + 3, lane), acc);
            int n = (nt - 8) * 16 + nl;
#pragma unroll
            for (int r = 0; r < 8; ++r) sB[(mbase + r) * 132 + n] = f2bf(acc[r]);
        }
        {                                          // nt = 16: n = 256..271
            v8f acc = {};
            acc = wmma_bf16(a0, load_bfrag(wIn, 16 * 4 + 0, lane), acc);
            acc = wmma_bf16(a1, load_bfrag(wIn, 16 * 4 + 1, lane), acc);
            acc = wmma_bf16(a2, load_bfrag(wIn, 16 * 4 + 2, lane), acc);
            acc = wmma_bf16(a3, load_bfrag(wIn, 16 * 4 + 3, lane), acc);
            int n = 256 + nl;
#pragma unroll
            for (int r = 0; r < 8; ++r) {
                float v = acc[r];
                int m = mbase + r;
                if (n < DINNER + CONVDIM)  sB[m * 132 + (n - DINNER)] = f2bf(v);
                else if (n < DINPROJ)      sDT[m * 8 + (n - DINNER - CONVDIM)] = v;
            }
        }
    }
    __syncthreads();

    // ---------- bidirectional mamba scan (D_STATE=1 -> scalar state per channel) ----------
#pragma unroll 1
    for (int dir = 0; dir < 2; ++dir) {
        const float* cw  = dir ? conv_w_b  : conv_w_f;
        const float* cb  = dir ? conv_b_b  : conv_b_f;
        const float* dtb = dir ? dt_bias_b : dt_bias_f;
        const float* alg = dir ? A_log_b   : A_log_f;
        const float* Dp  = dir ? D_b       : D_f;
        const float* nw  = dir ? norm_w_b  : norm_w_f;

        // pre-pass per (s,l): dt/dA per head, conv+silu for B,C channels
        if (tid < TOK_PER_WG) {
            int s = tid / LSEQ, l = tid % LSEQ;
            int t0 = tok_ord(s, l, dir);
            int t1 = (l >= 1) ? tok_ord(s, l - 1, dir) : -1;
            int t2 = (l >= 2) ? tok_ord(s, l - 2, dir) : -1;
#pragma unroll
            for (int c = 128; c < 130; ++c) {
                float u0 = bf2f(sB[t0 * 132 + c]);
                float u1 = (t1 >= 0) ? bf2f(sB[t1 * 132 + c]) : 0.f;
                float u2 = (t2 >= 0) ? bf2f(sB[t2 * 132 + c]) : 0.f;
                float v = cw[c * 3 + 0] * u2 + cw[c * 3 + 1] * u1 + cw[c * 3 + 2] * u0 + cb[c];
                sBC[(s * LSEQ + l) * 2 + (c - 128)] = silu_(v);
            }
#pragma unroll 1
            for (int h = 0; h < NHEADS; ++h) {
                float dt = softplus_(sDT[t0 * 8 + h] + dtb[h]);
                float A  = -__expf(alg[h]);
                float* d = &sdtA[((s * LSEQ + l) * NHEADS + h) * 2];
                d[0] = dt;
                d[1] = __expf(dt * A);
            }
        }
        __syncthreads();

        // per-channel scan: 8 seq * 128 ch = 1024 tasks over 256 threads
#pragma unroll 1
        for (int task = tid; task < SEQ_PER_WG * DINNER; task += 256) {
            int s = task >> 7, c = task & 127;
            int h = c >> 4;
            float w0 = cw[c * 3 + 0], w1 = cw[c * 3 + 1], w2 = cw[c * 3 + 2], b0 = cb[c];
            float Dh = Dp[h];
            float u1 = 0.f, u2 = 0.f, hstate = 0.f;
#pragma unroll 1
            for (int l = 0; l < LSEQ; ++l) {
                int tk = tok_ord(s, l, dir);
                float u0 = bf2f(sB[tk * 132 + c]);
                float xc = silu_(w0 * u2 + w1 * u1 + w2 * u0 + b0);
                const float* d = &sdtA[((s * LSEQ + l) * NHEADS + h) * 2];
                float Bc = sBC[(s * LSEQ + l) * 2 + 0];
                float Cc = sBC[(s * LSEQ + l) * 2 + 1];
                hstate = hstate * d[1] + d[0] * xc * Bc;
                float y = hstate * Cc + xc * Dh;
                y *= silu_(bf2f(sZ[tk * 128 + c]));
                sY[tk * 128 + c] = f2bf(y);
                u2 = u1; u1 = u0;
            }
        }
        __syncthreads();

        // RMS norm per token, accumulate 0.5*normed into sA
        if (tid < TOK_PER_WG) {
            float sq = 0.f;
#pragma unroll 1
            for (int j = 0; j < DINNER; ++j) { float v = bf2f(sY[tid * 128 + j]); sq += v * v; }
            float r = rsqrtf(sq * (1.f / 128.f) + EPSV);
#pragma unroll 1
            for (int j = 0; j < DINNER; ++j) {
                float v = 0.5f * bf2f(sY[tid * 128 + j]) * r * nw[j];
                if (dir == 0) sA[tid * 128 + j] = f2bf(v);
                else          sA[tid * 128 + j] = f2bf(bf2f(sA[tid * 128 + j]) + v);
            }
        }
        __syncthreads();
    }

    // ---------- reload x (residual) into sX1 ----------
    if (tid < TOK_PER_WG) {
        int b, t, p; tok2btp(blk, tid, b, t, p);
        int psrc = curves[b * PP + p];
        long long base = ((long long)(b * TT + t) * PP + psrc);
        const float* xr = x + base * DIMC;
#pragma unroll 1
        for (int j = 0; j < DIMC; ++j) sX1[tid * 96 + j] = xr[j];
    }
    __syncthreads();

    // ---------- out_proj GEMM: x1 = x + y_avg @ [128x96] ----------
#pragma unroll 1
    for (int mt = wave; mt < 15; mt += 8) {
        v16bf a0 = load_afrag(sA, mt * 16,  0, 128, lane);
        v16bf a1 = load_afrag(sA, mt * 16, 32, 128, lane);
        v16bf a2 = load_afrag(sA, mt * 16, 64, 128, lane);
        v16bf a3 = load_afrag(sA, mt * 16, 96, 128, lane);
#pragma unroll 1
        for (int nt = 0; nt < 6; ++nt) {
            v8f acc = {};
            acc = wmma_bf16(a0, load_bfrag(wOut, nt * 4 + 0, lane), acc);
            acc = wmma_bf16(a1, load_bfrag(wOut, nt * 4 + 1, lane), acc);
            acc = wmma_bf16(a2, load_bfrag(wOut, nt * 4 + 2, lane), acc);
            acc = wmma_bf16(a3, load_bfrag(wOut, nt * 4 + 3, lane), acc);
            int n = nt * 16 + (lane & 15);
            int mbase = mt * 16 + (lane >> 4) * 8;
#pragma unroll
            for (int r = 0; r < 8; ++r) sX1[(mbase + r) * 96 + n] += acc[r];
        }
    }
    __syncthreads();

    // ---------- LN(x1) + tanh(xs@loan2) -> z2 (bf16, sA cols 0..95) ----------
    if (tid < TOK_PER_WG) {
        float sum = 0.f, sq = 0.f;
#pragma unroll 1
        for (int j = 0; j < DIMC; ++j) { float v = sX1[tid * 96 + j]; sum += v; sq += v * v; }
        float mu = sum * (1.f / 96.f);
        float rstd = rsqrtf(sq * (1.f / 96.f) - mu * mu + EPSV);
#pragma unroll 1
        for (int j = 0; j < DIMC; ++j) {
            float a = loan2_b[j];
#pragma unroll
            for (int k = 0; k < DIMS; ++k) a += sXS[tid * 8 + k] * loan2_w[k * DIMC + j];
            float v = (sX1[tid * 96 + j] - mu) * rstd + tanhf(a);
            sA[tid * 128 + j] = f2bf(v);
        }
    }
    __syncthreads();

    // ---------- mlp1: gelu(z2 @ w1 + b1) -> sH1 (bf16) ----------
#pragma unroll 1
    for (int mt = wave; mt < 15; mt += 8) {
        v16bf a0 = load_afrag(sA, mt * 16,  0, 128, lane);
        v16bf a1 = load_afrag(sA, mt * 16, 32, 128, lane);
        v16bf a2 = load_afrag(sA, mt * 16, 64, 128, lane);
#pragma unroll 1
        for (int nt = 0; nt < 6; ++nt) {
            v8f acc = {};
            acc = wmma_bf16(a0, load_bfrag(wM1, nt * 3 + 0, lane), acc);
            acc = wmma_bf16(a1, load_bfrag(wM1, nt * 3 + 1, lane), acc);
            acc = wmma_bf16(a2, load_bfrag(wM1, nt * 3 + 2, lane), acc);
            int n = nt * 16 + (lane & 15);
            float b1 = mlp_b1[n];
            int mbase = mt * 16 + (lane >> 4) * 8;
#pragma unroll
            for (int r = 0; r < 8; ++r) {
                float v = acc[r] + b1;
                v = 0.5f * v * (1.f + erff(v * 0.70710678118f));   // exact gelu
                sH1[(mbase + r) * 96 + n] = f2bf(v);
            }
        }
    }
    __syncthreads();

    // ---------- mlp2 + final residual -> global out ----------
#pragma unroll 1
    for (int mt = wave; mt < 15; mt += 8) {
        v16bf a0 = load_afrag(sH1, mt * 16,  0, 96, lane);
        v16bf a1 = load_afrag(sH1, mt * 16, 32, 96, lane);
        v16bf a2 = load_afrag(sH1, mt * 16, 64, 96, lane);
#pragma unroll 1
        for (int nt = 0; nt < 6; ++nt) {
            v8f acc = {};
            acc = wmma_bf16(a0, load_bfrag(wM2, nt * 3 + 0, lane), acc);
            acc = wmma_bf16(a1, load_bfrag(wM2, nt * 3 + 1, lane), acc);
            acc = wmma_bf16(a2, load_bfrag(wM2, nt * 3 + 2, lane), acc);
            int n = nt * 16 + (lane & 15);
            float b2 = mlp_b2[n];
            int mbase = mt * 16 + (lane >> 4) * 8;
#pragma unroll
            for (int r = 0; r < 8; ++r) {
                int m = mbase + r;
                float v = sX1[m * 96 + n] + acc[r] + b2;
                int b, t, p; tok2btp(blk, m, b, t, p);
                out[(((long long)(b * TT + t)) * PP + p) * DIMC + n] = v;
            }
        }
    }
}

// ---------------- host launch ----------------
extern "C" void kernel_launch(void* const* d_in, const int* in_sizes, int n_in,
                              void* d_out, int out_size, void* d_ws, size_t ws_size,
                              hipStream_t stream) {
    (void)in_sizes; (void)n_in; (void)out_size; (void)ws_size;
    const float* x         = (const float*)d_in[0];
    const float* x_static  = (const float*)d_in[1];
    const float* x_hres    = (const float*)d_in[2];
    const int*   x_curves  = (const int*)d_in[3];
    const float* embed_w   = (const float*)d_in[4];
    const float* embed_b   = (const float*)d_in[5];
    const float* loan1_w   = (const float*)d_in[6];
    const float* loan1_b   = (const float*)d_in[7];
    const float* loan2_w   = (const float*)d_in[8];
    const float* loan2_b   = (const float*)d_in[9];
    const float* in_proj_w = (const float*)d_in[10];
    const float* conv_w_f  = (const float*)d_in[11];
    const float* conv_b_f  = (const float*)d_in[12];
    const float* dt_bias_f = (const float*)d_in[13];
    const float* A_log_f   = (const float*)d_in[14];
    const float* D_f       = (const float*)d_in[15];
    const float* norm_w_f  = (const float*)d_in[16];
    const float* conv_w_b  = (const float*)d_in[17];
    const float* conv_b_b  = (const float*)d_in[18];
    const float* dt_bias_b = (const float*)d_in[19];
    const float* A_log_b   = (const float*)d_in[20];
    const float* D_b       = (const float*)d_in[21];
    const float* norm_w_b  = (const float*)d_in[22];
    const float* out_proj_w= (const float*)d_in[23];
    const float* mlp_w1    = (const float*)d_in[24];
    const float* mlp_b1    = (const float*)d_in[25];
    const float* mlp_w2    = (const float*)d_in[26];
    const float* mlp_b2    = (const float*)d_in[27];

    unsigned short* wsp  = (unsigned short*)d_ws;
    unsigned short* wIn  = wsp;                       // 17*4 frags
    unsigned short* wOut = wIn  + (size_t)17 * 4 * 512;
    unsigned short* wM1  = wOut + (size_t)6  * 4 * 512;
    unsigned short* wM2  = wM1  + (size_t)6  * 3 * 512;

    pack_wmma_b<<<17 * 4, 32, 0, stream>>>(in_proj_w,  wIn,  128, 266, 4);
    pack_wmma_b<<<6 * 4,  32, 0, stream>>>(out_proj_w, wOut, 128, 96,  4);
    pack_wmma_b<<<6 * 3,  32, 0, stream>>>(mlp_w1,     wM1,  96,  96,  3);
    pack_wmma_b<<<6 * 3,  32, 0, stream>>>(mlp_w2,     wM2,  96,  96,  3);

    fused_mamba_block<<<NWG, 256, 0, stream>>>(
        x, x_static, x_hres, x_curves,
        embed_w, embed_b, loan1_w, loan1_b, loan2_w, loan2_b,
        conv_w_f, conv_b_f, dt_bias_f, A_log_f, D_f, norm_w_f,
        conv_w_b, conv_b_b, dt_bias_b, A_log_b, D_b, norm_w_b,
        mlp_b1, mlp_b2, wIn, wOut, wM1, wM2,
        (float*)d_out);
}
